// TransformerReformLM_26585847562608
// MI455X (gfx1250) — matrix operand order repaired and checked by reading, not compiled, not run on hardware
//
#include <hip/hip_runtime.h>
#include <hip/hip_bf16.h>

// ---------------------------------------------------------------------------
// Reformer forward for MI455X (gfx1250, wave32, WMMA + TDM).
// Dense math: v_wmma_f32_16x16x32_bf16 (f32 accumulate).
// GEMM weight tiles: tensor_load_to_lds (TDM) + s_wait_tensorcnt when the
// builtin is available; plain staging fallback otherwise.
// ---------------------------------------------------------------------------

#define RB      8
#define RS      512
#define RFDIM   512
#define RSUB    4
#define RDIM    128
#define RHEADS  8
#define RLOCALH 2
#define RLSHH   6
#define RDH     64
#define RINNER  512
#define RBUCKET 64
#define RNH     4
#define RMEM    128
#define RDEPTH  2
#define RT      2176              // S*SUB + MEM
#define RNB     34                // T / BUCKET
#define RC      136               // N_HASHES * NB (sorted chunks)
#define RWIN    128
#define RNW     17                // T / WINDOW
#define RSELF   (-5e4f)
#define RNEG    (-1e9f)

typedef __attribute__((ext_vector_type(16))) __bf16 v16bf;
typedef __attribute__((ext_vector_type(8)))  float  v8f;

__device__ __forceinline__ v8f wmma_bf16(v16bf a, v16bf b, v8f c) {
  return __builtin_amdgcn_wmma_f32_16x16x32_bf16(false, a, false, b, (short)0, c, false, false);
}

// pack two f32 into one dword of bf16 (lets staging use ds_store_b32)
__device__ __forceinline__ unsigned pk_bf16(float a, float b) {
  union { __bf16 h[2]; unsigned u; } u;
  u.h[0] = (__bf16)a; u.h[1] = (__bf16)b;
  return u.u;
}

// A fragment (16x32 bf16, M x K) from row-major bf16 LDS tile.
__device__ __forceinline__ v16bf frag_a_bf16(const __bf16* s, int row0, int ldk, int kBase) {
  int lane = threadIdx.x & 31;
  const __bf16* p = s + (size_t)(row0 + (lane & 15)) * ldk + kBase + ((lane >> 4) << 4);
  v16bf a;
#pragma unroll
  for (int e = 0; e < 16; ++e) a[e] = p[e];
  return a;
}

// A fragment built from row-major f32 LDS tile (converted on the fly).
__device__ __forceinline__ v16bf frag_a_f32(const float* s, int row0, int ldk, int kBase) {
  int lane = threadIdx.x & 31;
  const float* p = s + (size_t)(row0 + (lane & 15)) * ldk + kBase + ((lane >> 4) << 4);
  v16bf a;
#pragma unroll
  for (int e = 0; e < 16; ++e) a[e] = (__bf16)p[e];
  return a;
}

// B fragment (32x16 bf16, K x N) from row-major bf16 LDS tile [K][ldn].
__device__ __forceinline__ v16bf frag_b_bf16(const __bf16* s, int kBase, int ldn, int col0) {
  int lane = threadIdx.x & 31;
  const __bf16* p = s + (size_t)(kBase + ((lane >> 4) << 4)) * ldn + col0 + (lane & 15);
  v16bf b;
#pragma unroll
  for (int e = 0; e < 16; ++e) b[e] = p[(size_t)e * ldn];
  return b;
}

// B fragment from row-major f32 LDS tile [K][ldn] (TDM delivers raw f32).
__device__ __forceinline__ v16bf frag_b_f32(const float* s, int kBase, int ldn, int col0) {
  int lane = threadIdx.x & 31;
  const float* p = s + (size_t)(kBase + ((lane >> 4) << 4)) * ldn + col0 + (lane & 15);
  v16bf b;
#pragma unroll
  for (int e = 0; e < 16; ++e) b[e] = (__bf16)p[(size_t)e * ldn];
  return b;
}

__device__ __forceinline__ float wave_sum32(float v) {
#pragma unroll
  for (int off = 16; off; off >>= 1) v += __shfl_xor(v, off, 32);
  return v;
}

// ---------------------------------------------------------------------------
// Tensor Data Mover: 2D f32 tile (tileH rows x tileW cols) from a row-major
// [*, rowLen] tensor into contiguous LDS. Descriptor per CDNA5 ISA ch.8.
// This toolchain exposes the 6-arg builtin:
//   (uint32x4 g0, int32x8 g1, int32x4 g2, int32x4 g3, int32x8 extra, i32 cpol)
// ---------------------------------------------------------------------------
#if __has_builtin(__builtin_amdgcn_tensor_load_to_lds)
#define RF_HAVE_TDM 1
typedef __attribute__((ext_vector_type(4))) unsigned v4u;
typedef __attribute__((ext_vector_type(8))) int      v8i;
typedef __attribute__((ext_vector_type(4))) int      v4i;

__device__ __forceinline__ void rf_tdm_load_tile_f32(unsigned lds_byte_addr,
                                                     const float* gptr,
                                                     unsigned rowLen, unsigned nRows,
                                                     unsigned tileW, unsigned tileH) {
  unsigned long long ga = (unsigned long long)(size_t)gptr;
  v4u g0;
  g0.x = 1u;                                                   // count=1 (valid user D#)
  g0.y = lds_byte_addr;                                        // lds_addr
  g0.z = (unsigned)ga;                                         // global_addr[31:0]
  g0.w = ((unsigned)(ga >> 32) & 0x01FFFFFFu) | 0x80000000u;   // addr[56:32] | type=2
  v8i g1;
  g1[0] = (int)(2u << 16);                                     // data_size=4B, no mask
  g1[1] = (int)((rowLen & 0xFFFFu) << 16);                     // tensor_dim0 lo16
  g1[2] = (int)((rowLen >> 16) | ((nRows & 0xFFFFu) << 16));   // dim0 hi16 | dim1 lo16
  g1[3] = (int)((nRows >> 16) | (tileW << 16));                // dim1 hi16 | tile_dim0
  g1[4] = (int)tileH;                                          // tile_dim1 (tile_dim2=0)
  g1[5] = (int)rowLen;                                         // dim0_stride lo32
  g1[6] = 0;                                                   // stride hi / dim1_stride
  g1[7] = 0;
  v4i gz4 = {0, 0, 0, 0};
  v8i gz8 = {0, 0, 0, 0, 0, 0, 0, 0};
  __builtin_amdgcn_tensor_load_to_lds(g0, g1, gz4, gz4, gz8, 0);
}
#endif

// ---------------------------------------------------------------------------
// 0) positional encoding + subspace reshape (contiguous => same flat layout)
// ---------------------------------------------------------------------------
__global__ void rf_posenc(const float* __restrict__ feat, float* __restrict__ x1,
                          float* __restrict__ x2) {
  int idx = blockIdx.x * 256 + threadIdx.x;
  if (idx >= RB * RS * RFDIM) return;
  int j = idx & (RFDIM - 1);
  int s = (idx / RFDIM) % RS;
  float div = __powf(10000.0f, 2.0f * (float)j / (float)RFDIM);
  float ang = (float)s / div;
  float pe = (j & 1) ? __cosf(ang) : __sinf(ang);
  float v = feat[idx] + pe;
  x1[idx] = v;
  x2[idx] = v;
}

// ---------------------------------------------------------------------------
// 1) LayerNorm over rows of width 128 (one wave per row).
// ---------------------------------------------------------------------------
__global__ void rf_layernorm(const float* __restrict__ in, const float* __restrict__ in2,
                             const float* __restrict__ g, const float* __restrict__ bb,
                             float* __restrict__ out, int rows, int oSrcB, int oLogB) {
  int wid = blockIdx.x * 8 + (threadIdx.x >> 5);
  if (wid >= rows) return;
  int lane = threadIdx.x & 31;
  const float* p = in + (size_t)wid * RDIM;
  float v[4];
#pragma unroll
  for (int q = 0; q < 4; ++q) {
    float x = p[lane + 32 * q];
    if (in2) x = 0.5f * (x + in2[(size_t)wid * RDIM + lane + 32 * q]);
    v[q] = x;
  }
  float mean = wave_sum32(v[0] + v[1] + v[2] + v[3]) * (1.0f / RDIM);
  float ss = 0.f;
#pragma unroll
  for (int q = 0; q < 4; ++q) { float d = v[q] - mean; ss += d * d; }
  float var = wave_sum32(ss) * (1.0f / RDIM);
  float rstd = rsqrtf(var + 1e-5f);
  int orow = (oSrcB == oLogB) ? wid : (wid / oLogB) * oSrcB + (wid % oLogB);
#pragma unroll
  for (int q = 0; q < 4; ++q) {
    int j = lane + 32 * q;
    out[(size_t)orow * RDIM + j] = (v[q] - mean) * rstd * g[j] + bb[j];
  }
}

// mem_kv rows appended (raw, not layer-normed) to xm for each batch
__global__ void rf_fill_mem(const float* __restrict__ mem, float* __restrict__ xm) {
  int idx = blockIdx.x * 256 + threadIdx.x;
  if (idx >= RB * RMEM * RDIM) return;
  int f = idx & (RDIM - 1);
  int m = (idx / RDIM) % RMEM;
  int b = idx / (RDIM * RMEM);
  xm[((size_t)b * RT + (RS * RSUB) + m) * RDIM + f] = mem[m * RDIM + f];
}

__global__ void rf_zero(float* __restrict__ p, int n) {
  int idx = blockIdx.x * 256 + threadIdx.x;
  if (idx < n) p[idx] = 0.f;
}

// ---------------------------------------------------------------------------
// 2) bf16 WMMA GEMM:  C[M,N] = act( A[M,K] @ W[K,N] + bias ) + res
//    block tile 64x64, 8 waves. A staged as packed-bf16 dwords; W tile
//    staged as raw f32 via TDM (fallback: plain copy), converted at
//    fragment build. flags: 1=bias, 2=residual add, 4=exact GELU.
// ---------------------------------------------------------------------------
__global__ void rf_gemm_bf16(const float* __restrict__ A, const float* __restrict__ W,
                             const float* __restrict__ bias, const float* __restrict__ res,
                             float* __restrict__ C, int M, int N, int K,
                             int aSrcB, int aLogB, int flags) {
  __shared__ __bf16 As[64 * 32];   // A tile, bf16
  __shared__ float  Wf[32 * 64];   // W tile, raw f32 (TDM destination)
  int lane = threadIdx.x & 31, wid = threadIdx.x >> 5;
  int mBase = blockIdx.y * 64, nBase = blockIdx.x * 64;
  int mT = (wid & 3) * 16, nT = (wid >> 2) * 32;
  v8f c0 = {}; v8f c1 = {};
#if defined(RF_HAVE_TDM)
  unsigned wf_lds = (unsigned)(size_t)(void*)&Wf[0];  // low 32 bits = LDS byte offset
#endif
  for (int k0 = 0; k0 < K; k0 += 32) {
    // A tile: 64x32 f32 -> packed bf16 (1024 dword stores)
    for (int idx = threadIdx.x; idx < 1024; idx += 256) {
      int r = idx >> 4, kk = (idx & 15) * 2;
      int gr = mBase + r;
      int ar = (aSrcB == aLogB) ? gr : (gr / aLogB) * aSrcB + (gr % aLogB);
      const float* ap = &A[(size_t)ar * K + k0 + kk];
      ((unsigned*)As)[idx] = pk_bf16(ap[0], ap[1]);
    }
#if defined(RF_HAVE_TDM)
    if (wid == 0) {
      rf_tdm_load_tile_f32(wf_lds, &W[(size_t)k0 * N + nBase],
                           (unsigned)N, 32u, 64u, 32u);
      __builtin_amdgcn_s_wait_tensorcnt(0);
    }
#else
    for (int idx = threadIdx.x; idx < 2048; idx += 256) {
      int kr = idx >> 6, cc = idx & 63;
      Wf[idx] = W[(size_t)(k0 + kr) * N + nBase + cc];
    }
#endif
    if (k0 + 32 < K) {  // global_prefetch_b8 for next weight tile
      __builtin_prefetch(&W[(size_t)(k0 + 32 + (threadIdx.x & 31)) * N + nBase], 0, 0);
    }
    __syncthreads();
    v16bf a = frag_a_bf16(As, mT, 32, 0);
    v16bf b0 = frag_b_f32(Wf, 0, 64, nT);
    c0 = wmma_bf16(a, b0, c0);
    v16bf b1 = frag_b_f32(Wf, 0, 64, nT + 16);
    c1 = wmma_bf16(a, b1, c1);
    __syncthreads();
  }
  int half = lane >> 4, n0 = lane & 15;
#pragma unroll
  for (int r = 0; r < 8; ++r) {
    int row = mBase + mT + r + 8 * half;
#pragma unroll
    for (int t = 0; t < 2; ++t) {
      int col = nBase + nT + 16 * t + n0;
      float v = t ? c1[r] : c0[r];
      if (flags & 1) v += bias[col];
      if (flags & 4) v = 0.5f * v * (1.0f + erff(v * 0.70710678118f));
      if (flags & 2) v += res[(size_t)row * N + col];
      C[(size_t)row * N + col] = v;
    }
  }
}

// ---------------------------------------------------------------------------
// 3) LSH hashing: one wave per (bh, t); 4 hash rounds, argmax over [rv,-rv].
// ---------------------------------------------------------------------------
__global__ void rf_lsh_buckets(const float* __restrict__ qk, const float* __restrict__ rot,
                               int* __restrict__ buckets) {
  int wid = blockIdx.x * 8 + (threadIdx.x >> 5);
  if (wid >= (RB * RLSHH) * RT) return;
  int lane = threadIdx.x & 31;
  int bh = wid / RT, t = wid % RT;
  int b = bh / RLSHH, head = RLOCALH + (bh % RLSHH);
  const float* qrow = qk + ((size_t)(b * RT + t) * RINNER + head * RDH);
#pragma unroll
  for (int h = 0; h < RNH; ++h) {
    float acc = 0.f;
    if (lane < RNB / 2) {
      for (int f = 0; f < RDH; ++f) acc += qrow[f] * rot[(f * RNH + h) * (RNB / 2) + lane];
    }
    float bv; int bi;
    if (lane < RNB / 2) {
      if (acc >= -acc) { bv = acc;  bi = lane; }
      else             { bv = -acc; bi = lane + RNB / 2; }
    } else { bv = -3.4e38f; bi = 1 << 30; }
#pragma unroll
    for (int off = 16; off; off >>= 1) {
      float ov = __shfl_xor(bv, off, 32);
      int   oi = __shfl_xor(bi, off, 32);
      if (ov > bv || (ov == bv && oi < bi)) { bv = ov; bi = oi; }
    }
    if (lane == 0) buckets[((size_t)bh * RNH + h) * RT + t] = bi + h * RNB;
  }
}

// ---------------------------------------------------------------------------
// 4) stable counting sort per (bh, hash): perm[bh][h*T + r] = h*T + pos.
// ---------------------------------------------------------------------------
__global__ void rf_lsh_sort(const int* __restrict__ buckets, int* __restrict__ perm) {
  __shared__ int sb[RT];
  __shared__ int cnt[RNB], start[RNB];
  int bh = blockIdx.x >> 2, h = blockIdx.x & 3;
  const int* bsrc = buckets + ((size_t)bh * RNH + h) * RT;
  if (threadIdx.x < RNB) cnt[threadIdx.x] = 0;
  __syncthreads();
  for (int i = threadIdx.x; i < RT; i += 256) {
    int lb = bsrc[i] - h * RNB;
    sb[i] = lb;
    atomicAdd(&cnt[lb], 1);
  }
  __syncthreads();
  if (threadIdx.x == 0) {
    int acc = 0;
    for (int i = 0; i < RNB; ++i) { start[i] = acc; acc += cnt[i]; }
  }
  __syncthreads();
  if (threadIdx.x < RNB) {
    int bkt = threadIdx.x, k = start[bkt];
    int* dst = perm + (size_t)bh * (RNH * RT) + h * RT;
    for (int i = 0; i < RT; ++i)
      if (sb[i] == bkt) dst[k++] = h * RT + i;
  }
}

// ---------------------------------------------------------------------------
// 5) per-position logsumexp across the 4 hash rounds
// ---------------------------------------------------------------------------
__global__ void rf_lse_combine(const float* __restrict__ logits, float* __restrict__ lse_tot) {
  int idx = blockIdx.x * 256 + threadIdx.x;
  if (idx >= (RB * RLSHH) * RT) return;
  int bh = idx / RT, pos = idx % RT;
  float l[RNH], m = -3.4e38f;
#pragma unroll
  for (int h = 0; h < RNH; ++h) {
    l[h] = logits[((size_t)bh * RNH + h) * RT + pos];
    m = fmaxf(m, l[h]);
  }
  float s = 0.f;
#pragma unroll
  for (int h = 0; h < RNH; ++h) s += __expf(l[h] - m);
  lse_tot[idx] = m + __logf(s);
}

// ---------------------------------------------------------------------------
// 6) LSH chunk attention. Block per (bh, chunk). pass1: per-hash LSE.
//    pass2: p = exp(dots - LSE_total); o = p @ v, atomic-scatter to out.
// ---------------------------------------------------------------------------
__global__ void rf_lsh_attn(const float* __restrict__ qk, const float* __restrict__ v,
                            const int* __restrict__ perm, float* __restrict__ logits,
                            const float* __restrict__ lse_tot, float* __restrict__ out,
                            int pass) {
  extern __shared__ char smem[];
  __bf16* qA = (__bf16*)smem;             // [64][64]
  __bf16* kT = qA + 64 * 64;              // [64][128]  (dim-major normalized keys)
  __bf16* vB = kT + 64 * 128;             // [128][64]
  float*  dots = (float*)(vB + 128 * 64); // [64][128]
  __shared__ int   qpos[64], qflat[64], kpos[128];
  __shared__ float rowl[64];

  int bh = blockIdx.x / RC, c = blockIdx.x % RC;
  int b = bh / RLSHH, head = RLOCALH + bh % RLSHH;
  int tid = threadIdx.x, lane = tid & 31, wid = tid >> 5;
  const int* pp = perm + (size_t)bh * (RNH * RT);

  if (tid < 64) {
    int i = tid, flat = pp[c * 64 + i], pos = flat % RT;
    qpos[i] = pos; qflat[i] = flat;
    const float* r = qk + ((size_t)(b * RT + pos) * RINNER + head * RDH);
    for (int f = 0; f < RDH; f += 2)
      ((unsigned*)qA)[(i * 64 + f) >> 1] = pk_bf16(r[f], r[f + 1]);
  } else if (tid < 192) {
    int j = tid - 64;                       // 0..127: [self chunk | previous chunk]
    int cc = (j < 64) ? c : (c + RC - 1) % RC;
    int flat = pp[cc * 64 + (j & 63)], pos = flat % RT;
    kpos[j] = pos;
    const float* r = qk + ((size_t)(b * RT + pos) * RINNER + head * RDH);
    float ss = 0.f;
    for (int f = 0; f < RDH; ++f) { float x = r[f]; ss += x * x; }
    float inv = 1.0f / fmaxf(sqrtf(ss), 1e-12f);
    for (int f = 0; f < RDH; ++f) kT[f * 128 + j] = (__bf16)(r[f] * inv);
    if (pass == 2) {
      const float* vr = v + ((size_t)(b * RT + pos) * RINNER + head * RDH);
      for (int f = 0; f < RDH; f += 2)
        ((unsigned*)vB)[(j * 64 + f) >> 1] = pk_bf16(vr[f], vr[f + 1]);
    }
  }
  __syncthreads();

  // dots = q @ k^T (64x128, K=64): 32 tiles, 4 per wave
#pragma unroll
  for (int q = 0; q < 4; ++q) {
    int tileId = wid * 4 + q;
    int tm = (tileId & 3) * 16, tn = (tileId >> 2) * 16;
    v8f acc = {};
    acc = wmma_bf16(frag_a_bf16(qA, tm, 64, 0),  frag_b_bf16(kT, 0, 128, tn),  acc);
    acc = wmma_bf16(frag_a_bf16(qA, tm, 64, 32), frag_b_bf16(kT, 32, 128, tn), acc);
    int half = lane >> 4, n0 = lane & 15;
#pragma unroll
    for (int r = 0; r < 8; ++r) {
      int i = tm + r + 8 * half, j = tn + n0;
      float d = acc[r] * 0.125f;           // 1/sqrt(64)
      int qp = qpos[i], kp = kpos[j];
      if (qp < kp)       d = RNEG;         // causal
      else if (qp == kp) d = RSELF;        // self token
      dots[i * 128 + j] = d;
    }
  }
  __syncthreads();

  if (pass == 1) {
    if (tid < 64) {
      int i = tid;
      float m = -3.4e38f;
      for (int j = 0; j < 128; ++j) m = fmaxf(m, dots[i * 128 + j]);
      float s = 0.f;
      for (int j = 0; j < 128; ++j) s += __expf(dots[i * 128 + j] - m);
      float lse = m + __logf(s);
      int flat = qflat[i];
      logits[((size_t)bh * RNH + flat / RT) * RT + (flat % RT)] = lse;
    }
    return;
  }

  // pass 2: unnormalized softmax against cross-hash LSE
  if (tid < 64) rowl[tid] = lse_tot[(size_t)bh * RT + qpos[tid]];
  __syncthreads();
  for (int idx = tid; idx < 64 * 128; idx += 256) {
    int i = idx >> 7;
    dots[idx] = __expf(dots[idx] - rowl[i]);
  }
  __syncthreads();

  // o = p @ v (64x64, K=128): 16 tiles, 2 per wave
#pragma unroll
  for (int q = 0; q < 2; ++q) {
    int tileId = wid * 2 + q;
    int tm = (tileId & 3) * 16, tn = (tileId >> 2) * 16;
    v8f acc = {};
#pragma unroll
    for (int ks = 0; ks < 128; ks += 32)
      acc = wmma_bf16(frag_a_f32(dots, tm, 128, ks), frag_b_bf16(vB, ks, 64, tn), acc);
    int half = lane >> 4, n0 = lane & 15;
#pragma unroll
    for (int r = 0; r < 8; ++r) {
      int i = tm + r + 8 * half;
      atomicAdd(&out[((size_t)(b * RT) + qpos[i]) * RINNER + head * RDH + tn + n0], acc[r]);
    }
  }
}

// ---------------------------------------------------------------------------
// 7) local attention (shared-QK, look_backward=1). Block per 64-query
//    half-window; 256 keys (prev window + current window).
// ---------------------------------------------------------------------------
__global__ void rf_local_attn(const float* __restrict__ qk, const float* __restrict__ v,
                              float* __restrict__ out) {
  extern __shared__ char smem[];
  __bf16* qA = (__bf16*)smem;              // [64][64]
  __bf16* kT = qA + 64 * 64;               // [64][256]
  __bf16* vB = kT + 64 * 256;              // [256][64]
  float*  dots = (float*)(vB + 256 * 64);  // [64][256]

  int g = blockIdx.x;
  int bh = g / (RNW * 2), seg = g % (RNW * 2);
  int w = seg >> 1, hlf = seg & 1;
  int b = bh / RLOCALH, head = bh % RLOCALH;
  int t0 = w * RWIN + hlf * 64;
  int tid = threadIdx.x, lane = tid & 31, wid = tid >> 5;

  if (tid < 64) {
    int tq = t0 + tid;
    const float* r = qk + ((size_t)(b * RT + tq) * RINNER + head * RDH);
    for (int f = 0; f < RDH; f += 2)
      ((unsigned*)qA)[(tid * 64 + f) >> 1] = pk_bf16(r[f], r[f + 1]);
  }
  {
    int j = tid;                                        // 0..255 keys
    int tk = (j < RWIN) ? (w - 1) * RWIN + j : w * RWIN + (j - RWIN);
    if (tk >= 0) {
      const float* r = qk + ((size_t)(b * RT + tk) * RINNER + head * RDH);
      float ss = 0.f;
      for (int f = 0; f < RDH; ++f) { float x = r[f]; ss += x * x; }
      float inv = 1.0f / fmaxf(sqrtf(ss), 1e-12f);
      for (int f = 0; f < RDH; ++f) kT[f * 256 + j] = (__bf16)(r[f] * inv);
      const float* vr = v + ((size_t)(b * RT + tk) * RINNER + head * RDH);
      for (int f = 0; f < RDH; f += 2)
        ((unsigned*)vB)[(j * 64 + f) >> 1] = pk_bf16(vr[f], vr[f + 1]);
    } else {
      for (int f = 0; f < RDH; ++f) kT[f * 256 + j] = (__bf16)0.f;
      for (int f = 0; f < RDH; f += 2)
        ((unsigned*)vB)[(j * 64 + f) >> 1] = 0u;
    }
  }
  __syncthreads();

  // dots 64x256, K=64: 64 tiles, 8 per wave
#pragma unroll
  for (int q = 0; q < 8; ++q) {
    int tileId = wid * 8 + q;
    int tm = (tileId & 3) * 16, tn = (tileId >> 2) * 16;
    v8f acc = {};
    acc = wmma_bf16(frag_a_bf16(qA, tm, 64, 0),  frag_b_bf16(kT, 0, 256, tn),  acc);
    acc = wmma_bf16(frag_a_bf16(qA, tm, 64, 32), frag_b_bf16(kT, 32, 256, tn), acc);
    int half = lane >> 4, n0 = lane & 15;
#pragma unroll
    for (int r = 0; r < 8; ++r) {
      int i = tm + r + 8 * half, j = tn + n0;
      int tq = t0 + i;
      int tk = (j < RWIN) ? (w - 1) * RWIN + j : w * RWIN + (j - RWIN);
      float d = acc[r] * 0.125f;
      if (tk < 0)        d = RNEG;   // padded (no previous window)
      else if (tq == tk) d = RSELF;  // self token
      else if (tq < tk)  d = RNEG;   // causal
      dots[i * 256 + j] = d;
    }
  }
  __syncthreads();

  if (tid < 64) {
    int i = tid;
    float m = -3.4e38f;
    for (int j = 0; j < 256; ++j) m = fmaxf(m, dots[i * 256 + j]);
    float s = 0.f;
    for (int j = 0; j < 256; ++j) s += __expf(dots[i * 256 + j] - m);
    float invs = 1.0f / s;
    for (int j = 0; j < 256; ++j)
      dots[i * 256 + j] = __expf(dots[i * 256 + j] - m) * invs;
  }
  __syncthreads();

  // o = p @ v (64x64, K=256): 16 tiles, 2 per wave
#pragma unroll
  for (int q = 0; q < 2; ++q) {
    int tileId = wid * 2 + q;
    int tm = (tileId & 3) * 16, tn = (tileId >> 2) * 16;
    v8f acc = {};
#pragma unroll
    for (int ks = 0; ks < 256; ks += 32)
      acc = wmma_bf16(frag_a_f32(dots, tm, 256, ks), frag_b_bf16(vB, ks, 64, tn), acc);
    int half = lane >> 4, n0 = lane & 15;
#pragma unroll
    for (int r = 0; r < 8; ++r) {
      int i = tm + r + 8 * half;
      out[((size_t)(b * RT) + t0 + i) * RINNER + head * RDH + tn + n0] = acc[r];
    }
  }
}

// ---------------------------------------------------------------------------
// 8) final mean-pool over S after subspace-merge back to [B, S, FDIM]
// ---------------------------------------------------------------------------
__global__ void rf_meanpool(const float* __restrict__ x, float* __restrict__ out) {
  int idx = blockIdx.x * 256 + threadIdx.x;
  if (idx >= RB * RFDIM) return;
  int b = idx / RFDIM, j = idx % RFDIM;
  int sub = j >> 7, col = j & (RDIM - 1);
  float acc = 0.f;
  for (int s = 0; s < RS; ++s)
    acc += x[(size_t)(b * (RS * RSUB) + s * RSUB + sub) * RDIM + col];
  out[idx] = acc * (1.0f / RS);
}

// ---------------------------------------------------------------------------
// host-side launch
// ---------------------------------------------------------------------------
extern "C" void kernel_launch(void* const* d_in, const int* in_sizes, int n_in,
                              void* d_out, int out_size, void* d_ws, size_t ws_size,
                              hipStream_t stream) {
  (void)in_sizes; (void)n_in; (void)out_size; (void)ws_size;
  const float* features  = (const float*)d_in[0];
  const float* w_qk      = (const float*)d_in[1];
  const float* w_v       = (const float*)d_in[2];
  const float* mem_kv    = (const float*)d_in[3];
  const float* w_out     = (const float*)d_in[4];
  const float* b_out     = (const float*)d_in[5];
  const float* ln1_g     = (const float*)d_in[6];
  const float* ln1_b     = (const float*)d_in[7];
  const float* ff_w1     = (const float*)d_in[8];
  const float* ff_b1     = (const float*)d_in[9];
  const float* ff_w2     = (const float*)d_in[10];
  const float* ff_b2     = (const float*)d_in[11];
  const float* ln2_g     = (const float*)d_in[12];
  const float* ln2_b     = (const float*)d_in[13];
  const float* lnf_g     = (const float*)d_in[14];
  const float* lnf_b     = (const float*)d_in[15];
  const float* rotations = (const float*)d_in[16];
  float* outp = (float*)d_out;

  const size_t nTok  = (size_t)RB * RS * RSUB;       // 16384 token rows
  const size_t nX    = nTok * RDIM;
  const size_t nXm   = (size_t)RB * RT * RDIM;
  const size_t nQK   = (size_t)RB * RT * RINNER;
  const size_t nH    = nTok * RINNER;
  const size_t nLg   = (size_t)RB * RLSHH * RNH * RT;
  const size_t nLt   = (size_t)RB * RLSHH * RT;
  const size_t nPerm = (size_t)RB * RLSHH * RNH * RT;

  float* ws   = (float*)d_ws;
  float* x1   = ws; ws += nX;
  float* x2   = ws; ws += nX;
  float* xm   = ws; ws += nXm;
  float* qk   = ws; ws += nQK;
  float* vv   = ws; ws += nQK;
  float* aout = ws; ws += nQK;
  float* gin  = ws; ws += nX;
  float* hbuf = ws; ws += nH;
  float* logi = ws; ws += nLg;
  float* lset = ws; ws += nLt;
  int* buckets = (int*)ws;
  int* perm    = (int*)(buckets + nPerm);

  const int lshShm   = (64 * 64 + 64 * 128 + 128 * 64) * 2 + 64 * 128 * 4;   //  73,728 B
  const int localShm = (64 * 64 + 64 * 256 + 256 * 64) * 2 + 64 * 256 * 4;   // 139,264 B

  auto gemm = [&](const float* A, const float* W, const float* bias, const float* res,
                  float* C, int M, int N, int K, int aSrcB, int aLogB, int flags) {
    rf_gemm_bf16<<<dim3(N / 64, M / 64), 256, 0, stream>>>(A, W, bias, res, C, M, N, K,
                                                           aSrcB, aLogB, flags);
  };

  rf_posenc<<<(RB * RS * RFDIM) / 256, 256, 0, stream>>>(features, x1, x2);

  for (int l = 0; l < RDEPTH; ++l) {
    const float* Wqk = w_qk + (size_t)l * RDIM * RINNER;
    const float* Wv  = w_v  + (size_t)l * RDIM * RINNER;
    const float* Mem = mem_kv + (size_t)l * RMEM * RDIM;
    const float* Wo  = w_out + (size_t)l * RINNER * RDIM;
    const float* Bo  = b_out + (size_t)l * RDIM;
    const float* Rot = rotations + (size_t)l * RDH * RNH * (RNB / 2);

    // attention input: xm = [LN(x2) ; mem_kv]
    rf_layernorm<<<(int)(nTok / 8), 256, 0, stream>>>(x2, nullptr,
        ln1_g + l * RDIM, ln1_b + l * RDIM, xm, (int)nTok, RT, RS * RSUB);
    rf_fill_mem<<<(RB * RMEM * RDIM) / 256, 256, 0, stream>>>(Mem, xm);

    // projections (shared-QK and V)
    gemm(xm, Wqk, nullptr, nullptr, qk, RB * RT, RINNER, RDIM, 0, 0, 0);
    gemm(xm, Wv,  nullptr, nullptr, vv, RB * RT, RINNER, RDIM, 0, 0, 0);

    // LSH hashing + stable bucket sort
    rf_lsh_buckets<<<(RB * RLSHH * RT) / 8, 256, 0, stream>>>(qk, Rot, buckets);
    rf_lsh_sort<<<RB * RLSHH * RNH, 256, 0, stream>>>(buckets, perm);

    // attention heads
    rf_zero<<<(int)((nQK + 255) / 256), 256, 0, stream>>>(aout, (int)nQK);
    rf_local_attn<<<RB * RLOCALH * RNW * 2, 256, localShm, stream>>>(qk, vv, aout);
    rf_lsh_attn<<<RB * RLSHH * RC, 256, lshShm, stream>>>(qk, vv, perm, logi, lset, aout, 1);
    rf_lse_combine<<<(int)((nLt + 255) / 256), 256, 0, stream>>>(logi, lset);
    rf_lsh_attn<<<RB * RLSHH * RC, 256, lshShm, stream>>>(qk, vv, perm, logi, lset, aout, 2);

    // y1 = x1 + out[:, :2048] @ w_out + b_out   (A rows skip the mem tail)
    gemm(aout, Wo, Bo, x1, x1, (int)nTok, RDIM, RINNER, RT, RS * RSUB, 1 | 2);

    // y2 = x2 + FF(LN(y1))
    rf_layernorm<<<(int)(nTok / 8), 256, 0, stream>>>(x1, nullptr,
        ln2_g + l * RDIM, ln2_b + l * RDIM, gin, (int)nTok, 0, 0);
    gemm(gin, ff_w1 + (size_t)l * RDIM * RINNER, ff_b1 + (size_t)l * RINNER,
         nullptr, hbuf, (int)nTok, RINNER, RDIM, 0, 0, 1 | 4);
    gemm(hbuf, ff_w2 + (size_t)l * RINNER * RDIM, ff_b2 + (size_t)l * RDIM,
         x2, x2, (int)nTok, RDIM, RINNER, 0, 0, 1 | 2);
  }

  // final: LN(0.5*(x1+x2)) -> merge subspaces -> mean over S
  rf_layernorm<<<(int)(nTok / 8), 256, 0, stream>>>(x1, x2, lnf_g, lnf_b, gin, (int)nTok, 0, 0);
  rf_meanpool<<<(RB * RFDIM + 255) / 256, 256, 0, stream>>>(gin, outp);
}
